// HeteroGAT_FullEncoder_47828755808352
// MI455X (gfx1250) — matrix operand order repaired and checked by reading, not compiled, not run on hardware
//
#include <hip/hip_runtime.h>
#include <hip/hip_bf16.h>

#define NTX    100000
#define NADDR  120000
#define NEDGE  150000
#define HC     256
#define HID    64
#define EMB    128
#define F_TX   165
#define F_ADDR 56

typedef _Float16 v16h __attribute__((ext_vector_type(16)));
typedef _Float16 v4h  __attribute__((ext_vector_type(4)));
typedef _Float16 v2h  __attribute__((ext_vector_type(2)));
typedef float    v8f  __attribute__((ext_vector_type(8)));
typedef float    v4f  __attribute__((ext_vector_type(4)));

union FragH { v16h v; _Float16 e[16]; };

#define BK 32

// ---------------------------------------------------------------------------
// WMMA GEMM: C[M,N] = A[M,K] @ B[K,N] (+ bias[N]).  fp32 in/out, f16 compute.
// Block = 256 threads = 8 waves, tile TBM=32 x TBN, waves 2x4; each wave owns
// NI 16x16 v_wmma_f32_16x16x32_f16 tiles sharing ONE A fragment per k-step.
// B staging loads two adjacent K rows per thread so the transposed LDS write
// packs (k,k+1) into half2 ds_store_b32.  Software pipelined: next K-tile's
// global loads are issued before the current tile's WMMAs.
// GUARD=true handles ragged M/N/K (input projections only).
// ---------------------------------------------------------------------------
template <int TBN, int NI, bool GUARD>
__global__ __launch_bounds__(256)
void gemm_wmma(const float* __restrict__ A, const float* __restrict__ B,
               const float* __restrict__ bias, float* __restrict__ C,
               int M, int N, int K)
{
  constexpr int TBM = 32;
  constexpr int LDK = BK + 4;          // half units; row stride 72B (8B aligned)
  constexpr int BU  = TBN / 64;        // B micro-units per thread (2x v4f each)

  __shared__ _Float16 As[TBM][LDK];
  __shared__ _Float16 Bs[TBN][LDK];    // transposed: Bs[n][k]

  const int tid  = threadIdx.x;
  const int wave = tid >> 5;
  const int lane = tid & 31;
  const int m0 = blockIdx.y * TBM;
  const int n0 = blockIdx.x * TBN;
  const int wm = (wave >> 2) << 4;     // 0 or 16
  const int wn = (wave & 3) * (NI * 16);

  v4f abuf;
  v4f bbuf[BU][2];

  auto loadTiles = [&](int k0) {
    {
      const int idx = tid * 4;         // element in TBM x BK tile
      const int r = idx >> 5, c = idx & 31;
      const int gm = m0 + r, gk = k0 + c;
      if (!GUARD) {
        abuf = *(const v4f*)&A[(size_t)gm * K + gk];
      } else {
        v4f v = {0.f, 0.f, 0.f, 0.f};
        if (gm < M) {
          const float* p = &A[(size_t)gm * K];
          if (gk + 0 < K) v.x = p[gk + 0];
          if (gk + 1 < K) v.y = p[gk + 1];
          if (gk + 2 < K) v.z = p[gk + 2];
          if (gk + 3 < K) v.w = p[gk + 3];
        }
        abuf = v;
      }
    }
#pragma unroll
    for (int u = 0; u < BU; ++u) {
      const int unit = tid + u * 256;
      const int kp = (unit & 15) * 2;  // even k
      const int n  = (unit >> 4) * 4;
      const int gk = k0 + kp, gn = n0 + n;
      if (!GUARD) {
        bbuf[u][0] = *(const v4f*)&B[(size_t)gk * N + gn];
        bbuf[u][1] = *(const v4f*)&B[(size_t)(gk + 1) * N + gn];
      } else {
        v4f z = {0.f, 0.f, 0.f, 0.f};
        bbuf[u][0] = z; bbuf[u][1] = z;
#pragma unroll
        for (int rr = 0; rr < 2; ++rr) {
          if (gk + rr < K) {
            const float* p = &B[(size_t)(gk + rr) * N];
            v4f v = z;
            if (gn + 0 < N) v.x = p[gn + 0];
            if (gn + 1 < N) v.y = p[gn + 1];
            if (gn + 2 < N) v.z = p[gn + 2];
            if (gn + 3 < N) v.w = p[gn + 3];
            bbuf[u][rr] = v;
          }
        }
      }
    }
  };

  auto storeTiles = [&]() {
    {
      const int idx = tid * 4;
      const int r = idx >> 5, c = idx & 31;
      v4h pk = {(_Float16)abuf.x, (_Float16)abuf.y,
                (_Float16)abuf.z, (_Float16)abuf.w};
      *(v4h*)&As[r][c] = pk;                       // ds_store_b64
    }
#pragma unroll
    for (int u = 0; u < BU; ++u) {
      const int unit = tid + u * 256;
      const int kp = (unit & 15) * 2;
      const int n  = (unit >> 4) * 4;
      v4f r0 = bbuf[u][0], r1 = bbuf[u][1];
      v2h p0 = {(_Float16)r0.x, (_Float16)r1.x};   // (k,k+1) contiguous in Bs[n][k]
      v2h p1 = {(_Float16)r0.y, (_Float16)r1.y};
      v2h p2 = {(_Float16)r0.z, (_Float16)r1.z};
      v2h p3 = {(_Float16)r0.w, (_Float16)r1.w};
      *(v2h*)&Bs[n + 0][kp] = p0;                  // ds_store_b32 x4
      *(v2h*)&Bs[n + 1][kp] = p1;
      *(v2h*)&Bs[n + 2][kp] = p2;
      *(v2h*)&Bs[n + 3][kp] = p3;
    }
  };

  v8f acc[NI] = {};
  const int mr  = lane & 15;
  const int kha = (lane >> 4) << 3;    // A frag: lanes 16-31 start at K=8
  const int khb = (lane >> 4) << 4;    // unused: B frag indexes with kha, same K order as A

  loadTiles(0);
  for (int k0 = 0; k0 < K; k0 += BK) {
    __syncthreads();                   // previous compute done before overwrite
    storeTiles();
    __syncthreads();
    if (k0 + BK < K) loadTiles(k0 + BK);  // next tile in flight during WMMAs

    FragH fa;
#pragma unroll
    for (int j = 0; j < 8; ++j) {      // A 16x32 f16 VGPR layout (interleaved)
      const int kof = kha + ((j & 3) << 1) + ((j >> 2) << 4);
      fa.e[2 * j]     = As[wm + mr][kof];
      fa.e[2 * j + 1] = As[wm + mr][kof + 1];
    }
#pragma unroll
    for (int ni = 0; ni < NI; ++ni) {
      FragH fb;
#pragma unroll
      for (int j = 0; j < 16; ++j)     // B 32x16 f16: same interleaved K order as A
        fb.e[j] = Bs[wn + ni * 16 + mr][kha + (j & 7) + ((j >> 3) << 4)];
      acc[ni] = __builtin_amdgcn_wmma_f32_16x16x32_f16(false, fa.v, false, fb.v,
                                                       (short)0, acc[ni], false, false);
    }
  }

#pragma unroll
  for (int ni = 0; ni < NI; ++ni) {
    const int n = n0 + wn + ni * 16 + (lane & 15);
    const float bv = (bias != nullptr && (!GUARD || n < N)) ? bias[n] : 0.0f;
#pragma unroll
    for (int r = 0; r < 8; ++r) {
      // C/D layout: VGPR r -> M = r (lanes 0-15) / r+8 (lanes 16-31)
      const int m = m0 + wm + r + ((lane >> 4) << 3);
      if (!GUARD || (m < M && n < N))
        C[(size_t)m * N + n] = acc[ni][r] + bv;
    }
  }
}

// ---------------------------------------------------------------------------
// wv[k,h] = sum_c W[k, h*64+c] * avec[h*64+c]   (collapse attention dot into W)
// ---------------------------------------------------------------------------
__global__ void make_attvec_kernel(const float* __restrict__ W,
                                   const float* __restrict__ avec,
                                   float* __restrict__ out, int K)
{
  int i = blockIdx.x * blockDim.x + threadIdx.x;
  if (i >= K * 4) return;
  int k = i >> 2, h = i & 3;
  const float* wr = W + (size_t)k * HC + h * 64;
  const float* av = avec + h * 64;
  float s = 0.f;
#pragma unroll 8
  for (int c = 0; c < 64; ++c) s += wr[c] * av[c];
  out[i] = s;
}

// S[n,h] = X[n,:] @ wv[:,h]   (K is a multiple of 4 here: 64 or 256)
__global__ void scores_kernel(const float* __restrict__ X,
                              const float* __restrict__ wv,
                              float* __restrict__ S, int N, int K)
{
  int n = blockIdx.x * blockDim.x + threadIdx.x;
  if (n >= N) return;
  const v4f* x = (const v4f*)(X + (size_t)n * K);
  float s0 = 0.f, s1 = 0.f, s2 = 0.f, s3 = 0.f;
  for (int k4 = 0; k4 < (K >> 2); ++k4) {
    v4f xv = x[k4];
    const v4f* w = (const v4f*)(wv + k4 * 16);
    v4f w0 = w[0], w1 = w[1], w2 = w[2], w3 = w[3];
    s0 += xv.x * w0.x + xv.y * w1.x + xv.z * w2.x + xv.w * w3.x;
    s1 += xv.x * w0.y + xv.y * w1.y + xv.z * w2.y + xv.w * w3.y;
    s2 += xv.x * w0.z + xv.y * w1.z + xv.z * w2.z + xv.w * w3.z;
    s3 += xv.x * w0.w + xv.y * w1.w + xv.z * w2.w + xv.w * w3.w;
  }
  S[n * 4 + 0] = s0; S[n * 4 + 1] = s1;
  S[n * 4 + 2] = s2; S[n * 4 + 3] = s3;
}

// O[n,:] = b0 + b1  (HeteroConv sums two relation biases per dst type)
__global__ void init_bias2_kernel(float* __restrict__ O,
                                  const float* __restrict__ B0,
                                  const float* __restrict__ B1, int total)
{
  int i = blockIdx.x * blockDim.x + threadIdx.x;
  if (i < total) O[i] = B0[i & (HC - 1)] + B1[i & (HC - 1)];
}

__global__ void init_softmax_kernel(unsigned* __restrict__ mx,
                                    float* __restrict__ den, int cnt)
{
  int i = blockIdx.x * blockDim.x + threadIdx.x;
  if (i < cnt) { mx[i] = 0u; den[i] = 0.f; }
}

// Monotone float<->uint mapping for atomicMax-based segment max.
__device__ __forceinline__ unsigned fmap(float f) {
  unsigned u = __float_as_uint(f);
  return (u & 0x80000000u) ? ~u : (u | 0x80000000u);
}
__device__ __forceinline__ float funmap(unsigned u) {
  u = (u & 0x80000000u) ? (u & 0x7FFFFFFFu) : ~u;
  return __uint_as_float(u);
}

__device__ __forceinline__ void edge_pair(int e, int E, const int* s, const int* d,
                                          int& ss, int& dd) {
  if (e < E) { ss = s[e]; dd = d[e]; } else { ss = dd = e - E; }  // self loops
}

__global__ void edge_max_kernel(const int* __restrict__ src, const int* __restrict__ dst,
                                const float* __restrict__ AS, const float* __restrict__ AD,
                                unsigned* __restrict__ mx, int E, int loops)
{
  int e = blockIdx.x * blockDim.x + threadIdx.x;
  if (e >= E + loops) return;
  int s, d; edge_pair(e, E, src, dst, s, d);
#pragma unroll
  for (int h = 0; h < 4; ++h) {
    float v = AS[s * 4 + h] + AD[d * 4 + h];
    v = v > 0.f ? v : 0.2f * v;                 // leaky relu
    atomicMax(&mx[d * 4 + h], fmap(v));
  }
}

__global__ void edge_exp_kernel(const int* __restrict__ src, const int* __restrict__ dst,
                                const float* __restrict__ AS, const float* __restrict__ AD,
                                const unsigned* __restrict__ mx,
                                float* __restrict__ den, float* __restrict__ exb,
                                int E, int loops)
{
  int e = blockIdx.x * blockDim.x + threadIdx.x;
  if (e >= E + loops) return;
  int s, d; edge_pair(e, E, src, dst, s, d);
#pragma unroll
  for (int h = 0; h < 4; ++h) {
    float v = AS[s * 4 + h] + AD[d * 4 + h];
    v = v > 0.f ? v : 0.2f * v;
    float t = __expf(v - funmap(mx[d * 4 + h]));
    exb[(size_t)e * 4 + h] = t;
    atomicAdd(&den[d * 4 + h], t);
  }
}

// One 256-thread block per edge: thread t handles channel t (head = t/64).
__global__ __launch_bounds__(256)
void edge_scatter_kernel(const int* __restrict__ src, const int* __restrict__ dst,
                         const float* __restrict__ G, const float* __restrict__ exb,
                         const float* __restrict__ den, float* __restrict__ O, int E)
{
  int e = blockIdx.x, t = threadIdx.x;
  int s, d; edge_pair(e, E, src, dst, s, d);
  int h = t >> 6;
  float alpha = exb[(size_t)e * 4 + h] / (den[d * 4 + h] + 1e-16f);
  atomicAdd(&O[(size_t)d * HC + t], G[(size_t)s * HC + t] * alpha);
}

// LayerNorm (+optional residual) + ELU, one block per node row of 256.
__global__ __launch_bounds__(256)
void ln_res_elu_kernel(const float* __restrict__ O, const float* __restrict__ R,
                       const float* __restrict__ w, const float* __restrict__ b,
                       float* __restrict__ out, int useRes)
{
  __shared__ float red[256];
  int n = blockIdx.x, t = threadIdx.x;
  float x = O[(size_t)n * HC + t];

  red[t] = x; __syncthreads();
  for (int s = 128; s > 0; s >>= 1) { if (t < s) red[t] += red[t + s]; __syncthreads(); }
  float mean = red[0] * (1.f / 256.f);
  __syncthreads();
  float dx = x - mean;
  red[t] = dx * dx; __syncthreads();
  for (int s = 128; s > 0; s >>= 1) { if (t < s) red[t] += red[t + s]; __syncthreads(); }
  float var = red[0] * (1.f / 256.f);

  float y = dx * rsqrtf(var + 1e-5f) * w[t] + b[t];
  if (useRes) y += R[(size_t)n * HC + t];
  out[(size_t)n * HC + t] = y > 0.f ? y : (__expf(y) - 1.f);
}

// ---------------------------------------------------------------------------
extern "C" void kernel_launch(void* const* d_in, const int* in_sizes, int n_in,
                              void* d_out, int out_size, void* d_ws, size_t ws_size,
                              hipStream_t stream)
{
  (void)in_sizes; (void)n_in; (void)out_size; (void)ws_size;

  const float* x_tx    = (const float*)d_in[0];
  const float* x_addr  = (const float*)d_in[1];
  const int* ei_src[4] = { (const int*)d_in[2], (const int*)d_in[4],
                           (const int*)d_in[6], (const int*)d_in[8] };
  const int* ei_dst[4] = { (const int*)d_in[3], (const int*)d_in[5],
                           (const int*)d_in[7], (const int*)d_in[9] };
  const float* proj_tx_W = (const float*)d_in[10];
  const float* proj_tx_b = (const float*)d_in[11];
  const float* proj_ad_W = (const float*)d_in[12];
  const float* proj_ad_b = (const float*)d_in[13];
  const float* Wl[2]    = { (const float*)d_in[14], (const float*)d_in[18] };
  const float* asrcl[2] = { (const float*)d_in[15], (const float*)d_in[19] };
  const float* adstl[2] = { (const float*)d_in[16], (const float*)d_in[20] };
  const float* bl[2]    = { (const float*)d_in[17], (const float*)d_in[21] };
  const float* ln_tx_w = (const float*)d_in[22];
  const float* ln_tx_b = (const float*)d_in[23];
  const float* ln_ad_w = (const float*)d_in[24];
  const float* ln_ad_b = (const float*)d_in[25];
  const float* out_W   = (const float*)d_in[26];
  const float* out_b   = (const float*)d_in[27];

  char* base = (char*)d_ws;
  size_t off = 0;
  auto alloc = [&](size_t bytes) -> char* {
    char* p = base + off;
    off = (off + bytes + 255) & ~(size_t)255;
    return p;
  };
  float*    h_tx  = (float*)alloc((size_t)NTX * HC * 4);
  float*    h_ad  = (float*)alloc((size_t)NADDR * HC * 4);
  float*    o_tx  = (float*)alloc((size_t)NTX * HC * 4);
  float*    o_ad  = (float*)alloc((size_t)NADDR * HC * 4);
  float*    g     = (float*)alloc((size_t)NADDR * HC * 4);   // per-relation src transform
  float*    h1_tx = (float*)alloc((size_t)NTX * HID * 4);
  float*    h1_ad = (float*)alloc((size_t)NADDR * HID * 4);
  float*    s_src = (float*)alloc((size_t)NADDR * 4 * 4);
  float*    s_dst = (float*)alloc((size_t)NADDR * 4 * 4);
  unsigned* mx    = (unsigned*)alloc((size_t)NADDR * 4 * 4);
  float*    den   = (float*)alloc((size_t)NADDR * 4 * 4);
  float*    exb   = (float*)alloc((size_t)(NEDGE + NADDR) * 4 * 4);
  float*    wsv   = (float*)alloc((size_t)HC * 4 * 4);
  float*    wdv   = (float*)alloc((size_t)HC * 4 * 4);

  // Fast GEMMs: M%32==0, K%32==0; pick tile by N.
  auto gemm_fast = [&](const float* A, const float* B, const float* bias, float* C,
                       int M, int N, int K) {
    if ((N & 255) == 0) {
      dim3 grid(N / 256, M / 32);
      gemm_wmma<256, 4, false><<<grid, 256, 0, stream>>>(A, B, bias, C, M, N, K);
    } else {
      dim3 grid((N + 127) / 128, M / 32);
      gemm_wmma<128, 2, false><<<grid, 256, 0, stream>>>(A, B, bias, C, M, N, K);
    }
  };
  // Guarded GEMM: ragged K/N (input projections).
  auto gemm_guard = [&](const float* A, const float* B, const float* bias, float* C,
                        int M, int N, int K) {
    dim3 grid((N + 63) / 64, (M + 31) / 32);
    gemm_wmma<64, 1, true><<<grid, 256, 0, stream>>>(A, B, bias, C, M, N, K);
  };

  // Input projections (ragged K=165/56, N=64)
  gemm_guard(x_tx,   proj_tx_W, proj_tx_b, h1_tx, NTX,   HID, F_TX);
  gemm_guard(x_addr, proj_ad_W, proj_ad_b, h1_ad, NADDR, HID, F_ADDR);

  // Relation metadata: 0=tt, 1=aa, 2=ta, 3=at
  const int relNsrc[4]  = { NTX, NADDR, NTX,   NADDR };
  const int relNdst[4]  = { NTX, NADDR, NADDR, NTX };
  const int relLoops[4] = { NTX, NADDR, 0, 0 };        // self-loops for homogeneous
  const int relSrcTx[4] = { 1, 0, 1, 0 };
  const int relDstTx[4] = { 1, 0, 0, 1 };

  for (int layer = 0; layer < 2; ++layer) {
    const int K = layer ? HC : HID;
    const float* ftx = layer ? h_tx : h1_tx;
    const float* fad = layer ? h_ad : h1_ad;
    const float* W  = Wl[layer];
    const float* as = asrcl[layer];
    const float* ad = adstl[layer];
    const float* bb = bl[layer];

    init_bias2_kernel<<<(NTX * HC + 255) / 256, 256, 0, stream>>>(o_tx, bb + 0 * HC, bb + 3 * HC, NTX * HC);
    init_bias2_kernel<<<(NADDR * HC + 255) / 256, 256, 0, stream>>>(o_ad, bb + 1 * HC, bb + 2 * HC, NADDR * HC);

    for (int r = 0; r < 4; ++r) {
      const float* Wr   = W + (size_t)r * K * HC;
      const float* Asrc = relSrcTx[r] ? ftx : fad;
      const float* Adst = relDstTx[r] ? ftx : fad;
      float*       Odst = relDstTx[r] ? o_tx : o_ad;
      const int Nsrc = relNsrc[r], Ndst = relNdst[r], loops = relLoops[r];
      const int Et = NEDGE + loops;

      // Collapse attention vectors into W: a_{s,d} = x @ (W . a)
      make_attvec_kernel<<<(K * 4 + 255) / 256, 256, 0, stream>>>(Wr, as + r * HC, wsv, K);
      make_attvec_kernel<<<(K * 4 + 255) / 256, 256, 0, stream>>>(Wr, ad + r * HC, wdv, K);

      gemm_fast(Asrc, Wr, nullptr, g, Nsrc, HC, K);  // messages need full src transform
      scores_kernel<<<(Nsrc + 255) / 256, 256, 0, stream>>>(Asrc, wsv, s_src, Nsrc, K);
      scores_kernel<<<(Ndst + 255) / 256, 256, 0, stream>>>(Adst, wdv, s_dst, Ndst, K);

      init_softmax_kernel<<<(Ndst * 4 + 255) / 256, 256, 0, stream>>>(mx, den, Ndst * 4);
      edge_max_kernel<<<(Et + 255) / 256, 256, 0, stream>>>(ei_src[r], ei_dst[r], s_src, s_dst, mx, NEDGE, loops);
      edge_exp_kernel<<<(Et + 255) / 256, 256, 0, stream>>>(ei_src[r], ei_dst[r], s_src, s_dst, mx, den, exb, NEDGE, loops);
      edge_scatter_kernel<<<Et, 256, 0, stream>>>(ei_src[r], ei_dst[r], g, exb, den, Odst, NEDGE);
    }

    // LayerNorm + residual (layer 2 only: shapes match) + ELU
    ln_res_elu_kernel<<<NTX,   256, 0, stream>>>(o_tx, ftx, ln_tx_w, ln_tx_b, h_tx, layer);
    ln_res_elu_kernel<<<NADDR, 256, 0, stream>>>(o_ad, fad, ln_ad_w, ln_ad_b, h_ad, layer);
  }

  // Final projection -> d_out [NTX, EMB]
  gemm_fast(h_tx, out_W, out_b, (float*)d_out, NTX, EMB, HC);
}
